// ContrastiveLoss_12386685681710
// MI455X (gfx1250) — compile-verified, matrix operand order checked by reading, not run on hardware
//
#include <hip/hip_runtime.h>

typedef __attribute__((ext_vector_type(16))) _Float16 v16h;
typedef __attribute__((ext_vector_type(8)))  _Float16 v8h;
typedef __attribute__((ext_vector_type(8)))  float    v8f;

#define D      256
#define INV_T  2.0f    // 1 / TEMPERATURE
#define MAXC   2.0f    // |cos| <= 1 -> logits bounded by 2: fixed softmax offset
#define SLICES 8       // column-slice blocks per row stripe
#define PH     264     // padded LDS row pitch (halves): 528B stride -> bank-conflict-free frags

// ---------------------------------------------------------------------------
// Phase 1: L2-normalize rows of concat(embeddings_i, embeddings_j) -> f16 z
// ---------------------------------------------------------------------------
__global__ __launch_bounds__(256) void nrm_kernel(const float* __restrict__ ei,
                                                  const float* __restrict__ ej,
                                                  _Float16* __restrict__ zh, int B) {
  const int lane = threadIdx.x & 31;
  const int wid  = threadIdx.x >> 5;
  const int row  = blockIdx.x * 8 + wid;
  const float* src = (row < B) ? (ei + (size_t)row * D)
                               : (ej + (size_t)(row - B) * D);
  float4 a = *(const float4*)(src + lane * 8);
  float4 b = *(const float4*)(src + lane * 8 + 4);
  float ss = a.x*a.x + a.y*a.y + a.z*a.z + a.w*a.w
           + b.x*b.x + b.y*b.y + b.z*b.z + b.w*b.w;
  #pragma unroll
  for (int m = 16; m >= 1; m >>= 1) ss += __shfl_xor(ss, m, 32);
  const float inv = 1.0f / sqrtf(ss);
  v8h o;
  o[0] = (_Float16)(a.x * inv); o[1] = (_Float16)(a.y * inv);
  o[2] = (_Float16)(a.z * inv); o[3] = (_Float16)(a.w * inv);
  o[4] = (_Float16)(b.x * inv); o[5] = (_Float16)(b.y * inv);
  o[6] = (_Float16)(b.z * inv); o[7] = (_Float16)(b.w * inv);
  *(v8h*)(zh + (size_t)row * D + lane * 8) = o;
}

// ---------------------------------------------------------------------------
// Fragment loaders for v_wmma_f32_16x16x32_f16 (gfx1250 wave32 layout).
// lanes 0-15 hold M(or N)=lane, K = {0..7, 16..23};
// lanes 16-31 hold M(or N)=lane-16, K = {8..15, 24..31}.
// ---------------------------------------------------------------------------
__device__ inline v16h load_afrag(const _Float16* __restrict__ z,
                                  int row0, int k0, int lane) {
  const int m  = lane & 15;
  const int kb = (lane < 16) ? 0 : 8;
  const _Float16* p = z + (size_t)(row0 + m) * D + (k0 + kb);
  v8h lo = *(const v8h*)(p);
  v8h hi = *(const v8h*)(p + 16);
  return __builtin_shufflevector(lo, hi, 0,1,2,3,4,5,6,7,8,9,10,11,12,13,14,15);
}

__device__ inline v16h lds_bfrag(const _Float16* buf, int k0, int lane) {
  const int n  = lane & 15;                 // B column index = staged z row
  const int kb = (lane < 16) ? 0 : 8;
  const _Float16* p = buf + n * PH + (k0 + kb);
  v8h lo = *(const v8h*)(p);                // ds_load_b128
  v8h hi = *(const v8h*)(p + 16);           // ds_load_b128
  return __builtin_shufflevector(lo, hi, 0,1,2,3,4,5,6,7,8,9,10,11,12,13,14,15);
}

// ---------------------------------------------------------------------------
// Phase 2: fused GEMM + softmax-denominator streaming.
// Block = 128 rows (one 16-row tile per wave) x 1/8 of the columns.
// B tiles double-buffered through LDS: loaded once per block, used by 8 waves
// (8x less L2 traffic than per-wave global loads).
// ---------------------------------------------------------------------------
__global__ __launch_bounds__(256) void simloss_kernel(const _Float16* __restrict__ zh,
                                                      float* __restrict__ ws2,
                                                      int N) {
  __shared__ _Float16 bbuf[2][16 * PH];     // 2 x 8.25 KB, contiguous
  const int tid    = threadIdx.x;
  const int lane   = tid & 31;
  const int wid    = tid >> 5;
  const int stripe = blockIdx.x >> 3;       // 128-row stripe
  const int slice  = blockIdx.x & (SLICES - 1);
  const int nTiles = N >> 4;
  const int tps    = nTiles / SLICES;       // tiles per slice
  const int tileBeg = slice * tps;
  const int half   = N >> 1;

  const int r0 = stripe * 128 + wid * 16;   // this wave's 16 rows

  // A fragments for the whole K range, resident in 64 VGPRs.
  v16h afrag[8];
  #pragma unroll
  for (int k = 0; k < 8; ++k) afrag[k] = load_afrag(zh, r0, k * 32, lane);

  // Staging: each of the 256 threads copies 32B of the 8KB tile.
  // Single LDS-derived base + integer buffer offset keeps addrspace(3)
  // (a pointer select would force FLAT stores).
  const int srow = tid >> 4;                // tile row 0..15
  const int scol = (tid & 15) * 16;         // halves 0..240
  const _Float16* gcol  = zh + scol;
  _Float16*       lbase = &bbuf[0][srow * PH + scol];
  const int       lstride = 16 * PH;        // halves between the two buffers

  float s[8], p[8];
  #pragma unroll
  for (int v = 0; v < 8; ++v) { s[v] = 0.0f; p[v] = 0.0f; }

  const int rbase = r0 + ((lane >> 4) << 3);
  const int cl    = lane & 15;

  // Prologue: stage tile 0 into buffer 0.
  {
    const _Float16* g = gcol + (size_t)(tileBeg * 16 + srow) * D;
    v8h a0 = *(const v8h*)(g);
    v8h a1 = *(const v8h*)(g + 8);
    *(v8h*)(lbase)     = a0;                // ds_store_b128
    *(v8h*)(lbase + 8) = a1;
  }
  __syncthreads();

  for (int i = 0; i < tps; ++i) {
    const bool haveNext = (i + 1) < tps;
    v8h n0v, n1v;
    if (haveNext) {                         // global loads overlap compute below
      const _Float16* g = gcol + (size_t)((tileBeg + i + 1) * 16 + srow) * D;
      n0v = *(const v8h*)(g);
      n1v = *(const v8h*)(g + 8);
    }

    const _Float16* buf = &bbuf[0][0] + (i & 1) * lstride;
    v8f acc = {0.f, 0.f, 0.f, 0.f, 0.f, 0.f, 0.f, 0.f};
    #pragma unroll
    for (int k = 0; k < 8; ++k) {
      v16h b = lds_bfrag(buf, k * 32, lane);
      acc = __builtin_amdgcn_wmma_f32_16x16x32_f16(false, afrag[k], false, b,
                                                   (short)0, acc, false, false);
    }

    const int n0  = (tileBeg + i) << 4;
    const int col = n0 + cl;
    #pragma unroll
    for (int v = 0; v < 8; ++v) {
      const int   row = rbase + v;
      const float val = acc[v] * INV_T;
      if (col != row) s[v] += __expf(val - MAXC);       // diag excluded
      const int label = (row < half) ? row + half : row - half;
      if (col == label) p[v] += val;                    // fires at most once
    }

    if (haveNext) {
      __syncthreads();                      // everyone done reading buf[(i+1)&1]
      _Float16* l = lbase + ((i + 1) & 1) * lstride;
      *(v8h*)(l)     = n0v;                 // ds_store_b128
      *(v8h*)(l + 8) = n1v;
      __syncthreads();                      // tile i+1 visible
    }
  }

  // Combine the 16 lanes that share each row (xor masks < 16 stay in-group).
  #pragma unroll
  for (int v = 0; v < 8; ++v) {
    #pragma unroll
    for (int m = 8; m >= 1; m >>= 1) {
      s[v] += __shfl_xor(s[v], m, 32);
      p[v] += __shfl_xor(p[v], m, 32);
    }
  }

  // Per-row (sumexp, positive) partials for this column slice; fixed slots.
  if (cl == 0) {
    #pragma unroll
    for (int v = 0; v < 8; ++v) {
      const int row = rbase + v;
      float2 sp; sp.x = s[v]; sp.y = p[v];
      *(float2*)&ws2[((size_t)slice * N + row) * 2] = sp;
    }
  }
}

// ---------------------------------------------------------------------------
// Phase 3: combine slices per row (fixed order), per-row loss, mean.
// ---------------------------------------------------------------------------
__global__ __launch_bounds__(256) void finalize_kernel(const float* __restrict__ ws2,
                                                       float* __restrict__ out,
                                                       int N) {
  __shared__ float red[256];
  float acc = 0.0f;
  for (int row = threadIdx.x; row < N; row += 256) {
    float S = 0.0f, P = 0.0f;
    #pragma unroll
    for (int sl = 0; sl < SLICES; ++sl) {
      float2 sp = *(const float2*)&ws2[((size_t)sl * N + row) * 2];
      S += sp.x; P += sp.y;
    }
    acc += logf(S) + MAXC - P;
  }
  red[threadIdx.x] = acc;
  __syncthreads();
  for (int st = 128; st > 0; st >>= 1) {
    if (threadIdx.x < st) red[threadIdx.x] += red[threadIdx.x + st];
    __syncthreads();
  }
  if (threadIdx.x == 0) out[0] = red[0] / (float)N;
}

extern "C" void kernel_launch(void* const* d_in, const int* in_sizes, int n_in,
                              void* d_out, int out_size, void* d_ws, size_t ws_size,
                              hipStream_t stream) {
  const float* ei = (const float*)d_in[0];   // embeddings_i [B, 256] f32
  const float* ej = (const float*)d_in[1];   // embeddings_j [B, 256] f32
  const int B = in_sizes[0] / D;             // 4096
  const int N = 2 * B;                       // 8192

  _Float16* zh  = (_Float16*)d_ws;                                    // 4 MB
  float*    ws2 = (float*)((char*)d_ws + (size_t)N * D * sizeof(_Float16));
                                             // SLICES * N * 2 floats = 512 KB
  float* out = (float*)d_out;

  nrm_kernel     <<<N / 8,              256, 0, stream>>>(ei, ej, zh, B);
  simloss_kernel <<<(N / 128) * SLICES, 256, 0, stream>>>(zh, ws2, N);
  finalize_kernel<<<1,                  256, 0, stream>>>(ws2, out, N);
}
